// Global_GNN_14336600834198
// MI455X (gfx1250) — matrix operand order repaired and checked by reading, compiled, not run on hardware
//
#include <hip/hip_runtime.h>
#include <hip/hip_bf16.h>
#include <math.h>

// ---------------------------------------------------------------------------
// Types for WMMA fragments (wave32, gfx1250)
// ---------------------------------------------------------------------------
typedef __attribute__((ext_vector_type(16))) __bf16 v16bf;
typedef __attribute__((ext_vector_type(16))) unsigned short v16us;
typedef __attribute__((ext_vector_type(8)))  float v8f;
// 16-byte-aligned view of a 32-byte ushort vector (LDS rows are 16B aligned)
typedef v16us v16us_a16 __attribute__((aligned(16)));

static __device__ __forceinline__ void split_hi_lo(float x, unsigned short& hi, unsigned short& lo) {
    unsigned u  = __float_as_uint(x);
    unsigned uh = u & 0xffff0000u;           // truncate mantissa -> exact bf16 "hi"
    hi = (unsigned short)(uh >> 16);
    float fl = x - __uint_as_float(uh);      // exact residual in f32
    lo = (unsigned short)(__float_as_uint(fl) >> 16);
}

// gfx1250 async global->LDS copy (VGLOBAL op 98): per-lane 16B, tracked by ASYNCcnt.
static __device__ __forceinline__ void async_ld_b128(unsigned lds_byte_off, const void* gptr) {
    asm volatile("global_load_async_to_lds_b128 %0, %1, off"
                 :: "v"(lds_byte_off), "v"((unsigned long long)(uintptr_t)gptr)
                 : "memory");
}

// ---------------------------------------------------------------------------
// Small utility kernels
// ---------------------------------------------------------------------------
__global__ void fill_f32(float* __restrict__ p, float v, int n) {
    int i = blockIdx.x * blockDim.x + threadIdx.x;
    if (i < n) p[i] = v;
}

__global__ void deg_count(const int* __restrict__ src, float* __restrict__ deg, int ne) {
    int e = blockIdx.x * blockDim.x + threadIdx.x;
    if (e < ne) atomicAdd(&deg[src[e]], 1.0f);
}

__global__ void deg_to_dis(float* __restrict__ dd, int n) {
    int i = blockIdx.x * blockDim.x + threadIdx.x;
    if (i < n) {
        float dg = dd[i];
        dd[i] = (dg > 0.0f) ? rsqrtf(fmaxf(dg, 1.0f)) : 0.0f;
    }
}

__global__ void norm_coef(const int* __restrict__ src, const int* __restrict__ dst,
                          const float* __restrict__ dis, float* __restrict__ nrm, int ne) {
    int e = blockIdx.x * blockDim.x + threadIdx.x;
    if (e < ne) nrm[e] = -dis[src[e]] * dis[dst[e]];
}

// Pre-transform W (shape (3, Kin, 20) row-major) into WMMA-B fragment order:
// Wt[kt*2048 + n*32 + kp] = W[kk= n/20][k = kt*32+kp][h = n%20], zero padded.
__global__ void wt_transform(const float* __restrict__ W, int Kin,
                             unsigned short* __restrict__ WtH,
                             unsigned short* __restrict__ WtL, int nkt) {
    int id = blockIdx.x * blockDim.x + threadIdx.x;
    int total = nkt * 2048;
    if (id >= total) return;
    int kt  = id >> 11;
    int rem = id & 2047;
    int n   = rem >> 5;
    int kp  = rem & 31;
    int k   = kt * 32 + kp;
    float v = 0.0f;
    if (k < Kin && n < 60) {
        int kk = n / 20, h = n - kk * 20;
        v = W[((size_t)kk * Kin + k) * 20 + h];
    }
    unsigned short h16, l16;
    split_hi_lo(v, h16, l16);
    WtH[id] = h16;
    WtL[id] = l16;
}

// ---------------------------------------------------------------------------
// Split-bf16 WMMA GEMM:  P(nrows x 64) = X(nrows x kvalid) @ Wcat(kvalid x 64)
// 3-term:  Ah*Bh + Ah*Bl + Al*Bh  accumulated in f32 (near-f32 accuracy).
// Block = 256 threads (8 waves), tile 128(M) x 64(N), K-step 32.
// B tiles streamed global->LDS with double-buffered async copies (ASYNCcnt).
// ---------------------------------------------------------------------------
__global__ __launch_bounds__(256)
void wmma_gemm_bf16x3(const float* __restrict__ X, int ldx, int kvalid,
                      int kt_full, int kt_total,
                      const unsigned short* __restrict__ WtH,
                      const unsigned short* __restrict__ WtL,
                      float* __restrict__ P, int nrows) {
    // Double-buffered padded B tiles: 64 cols x 32 k, row stride 40 ushorts
    // -> per-lane 32B fragment reads land on banks 0,4,...,60 (conflict-free b128).
    __shared__ __attribute__((aligned(16))) unsigned short ldsH[2][64 * 40];
    __shared__ __attribute__((aligned(16))) unsigned short ldsL[2][64 * 40];

    const int tid   = threadIdx.x;
    const int wave  = tid >> 5;
    const int lane  = tid & 31;
    const int lhalf = lane >> 4;   // 0 or 1
    const int l15   = lane & 15;

    const int rowTile = blockIdx.x * 128 + wave * 16;
    int rowA = rowTile + l15;
    if (rowA >= nrows) rowA = nrows - 1;           // clamp loads, guard stores
    const float* __restrict__ xr = X + (size_t)rowA * ldx;
    const int khalf = lhalf * 16;

    // B staging: thread tid moves 16B of H and 16B of L per tile.
    const int stg_n = tid >> 2;
    const int stg_k = (tid & 3) << 3;
    const unsigned stgH[2] = {
        (unsigned)(uintptr_t)&ldsH[0][stg_n * 40 + stg_k],
        (unsigned)(uintptr_t)&ldsH[1][stg_n * 40 + stg_k] };
    const unsigned stgL[2] = {
        (unsigned)(uintptr_t)&ldsL[0][stg_n * 40 + stg_k],
        (unsigned)(uintptr_t)&ldsL[1][stg_n * 40 + stg_k] };

    v8f acc[4];
    #pragma unroll
    for (int nt = 0; nt < 4; ++nt)
        #pragma unroll
        for (int r = 0; r < 8; ++r) acc[nt][r] = 0.0f;

    // Prologue: kick off tile 0 into buffer 0.
    async_ld_b128(stgH[0], (const uint4*)(WtH) + tid);
    async_ld_b128(stgL[0], (const uint4*)(WtL) + tid);

    for (int kt = 0; kt < kt_total; ++kt) {
        const int buf = kt & 1;
        // Issue tile kt+1 into the other buffer (freed by the barrier that ended
        // iteration kt-1), then wait for tile kt's two (older, in-order) copies.
        if (kt + 1 < kt_total) {
            const int nb = buf ^ 1;
            async_ld_b128(stgH[nb], (const uint4*)(WtH + (size_t)(kt + 1) * 2048) + tid);
            async_ld_b128(stgL[nb], (const uint4*)(WtL + (size_t)(kt + 1) * 2048) + tid);
            asm volatile("s_wait_asynccnt 0x2" ::: "memory");
        } else {
            asm volatile("s_wait_asynccnt 0x0" ::: "memory");
        }
        __syncthreads();   // tile kt visible to all waves

        // --- A fragment: lane holds row M=l15, K runs [k0,k0+8) and [k1,k1+8)
        v16us ah, al;
        const int k0 = kt * 32 + khalf;
        const int k1 = k0 + 8;
        if (kt < kt_full) {
            float va[16];
            float4 t;
            t = *(const float4*)(xr + k0);     va[0]=t.x;  va[1]=t.y;  va[2]=t.z;  va[3]=t.w;
            t = *(const float4*)(xr + k0 + 4); va[4]=t.x;  va[5]=t.y;  va[6]=t.z;  va[7]=t.w;
            t = *(const float4*)(xr + k1);     va[8]=t.x;  va[9]=t.y;  va[10]=t.z; va[11]=t.w;
            t = *(const float4*)(xr + k1 + 4); va[12]=t.x; va[13]=t.y; va[14]=t.z; va[15]=t.w;
            __builtin_prefetch(xr + k0 + 32, 0, 0);   // next K-tile -> global_prefetch_b8
            #pragma unroll
            for (int i = 0; i < 16; ++i) {
                unsigned short h16, l16;
                split_hi_lo(va[i], h16, l16);
                ah[i] = h16; al[i] = l16;
            }
        } else {  // guarded tail tile (also the only tile for the H=20 layer GEMMs)
            #pragma unroll
            for (int i = 0; i < 16; ++i) {
                int k = (i < 8) ? (k0 + i) : (k1 + i - 8);
                float v = (k < kvalid) ? xr[k] : 0.0f;
                unsigned short h16, l16;
                split_hi_lo(v, h16, l16);
                ah[i] = h16; al[i] = l16;
            }
        }
        const v16bf Ah = __builtin_bit_cast(v16bf, ah);
        const v16bf Al = __builtin_bit_cast(v16bf, al);

        // --- 4 N-tiles, 3 WMMA each
        #pragma unroll
        for (int nt = 0; nt < 4; ++nt) {
            const int col = nt * 16 + l15;
            const int bo  = col * 40 + lhalf * 16;       // 16 contiguous ushorts
            v16us bh = *(const v16us_a16*)(&ldsH[buf][bo]);
            v16us bl = *(const v16us_a16*)(&ldsL[buf][bo]);
            const v16bf Bh = __builtin_bit_cast(v16bf, bh);
            const v16bf Bl = __builtin_bit_cast(v16bf, bl);
            acc[nt] = __builtin_amdgcn_wmma_f32_16x16x32_bf16(false, Ah, false, Bh, (short)0, acc[nt], false, false);
            acc[nt] = __builtin_amdgcn_wmma_f32_16x16x32_bf16(false, Ah, false, Bl, (short)0, acc[nt], false, false);
            acc[nt] = __builtin_amdgcn_wmma_f32_16x16x32_bf16(false, Al, false, Bh, (short)0, acc[nt], false, false);
        }
        __syncthreads();   // all waves done reading buf before it is re-staged
    }

    // --- store D: element r of acc <-> row (lhalf*8 + r), col (nt*16 + l15)
    const int mrow0 = rowTile + lhalf * 8;
    #pragma unroll
    for (int nt = 0; nt < 4; ++nt) {
        const int col = nt * 16 + l15;
        #pragma unroll
        for (int r = 0; r < 8; ++r) {
            int row = mrow0 + r;
            if (row < nrows) P[(size_t)row * 64 + col] = acc[nt][r];
        }
    }
}

// ---------------------------------------------------------------------------
// Graph propagation: y[dst] += norm * x[src]   (edge-parallel, f32 atomics)
// ---------------------------------------------------------------------------
__global__ void prop_scatter(const float* __restrict__ x, int ldx,
                             float* __restrict__ y, int ldy, int ncols,
                             const int* __restrict__ src, const int* __restrict__ dst,
                             const float* __restrict__ nrm, int ne) {
    int e = blockIdx.x * blockDim.x + threadIdx.x;
    if (e >= ne) return;
    int s = src[e], d = dst[e];
    float wv = nrm[e];
    const float* xs = x + (size_t)s * ldx;
    float* yd = y + (size_t)d * ldy;
    for (int c = 0; c < ncols; c += 4) {
        float4 v = *(const float4*)(xs + c);
        atomicAdd(yd + c + 0, wv * v.x);
        atomicAdd(yd + c + 1, wv * v.y);
        atomicAdd(yd + c + 2, wv * v.z);
        atomicAdd(yd + c + 3, wv * v.w);
    }
}

// ---------------------------------------------------------------------------
// Cheb epilogue + BN + ReLU + residual:
//   pre = p0 + t1 + 2*t2 - p2 + b ;  y = relu(bn(pre)) [+ 0.7*Lprev]
// ---------------------------------------------------------------------------
__global__ void cheb_post(const float* __restrict__ P, const float* __restrict__ T,
                          const float* __restrict__ T2, const float* __restrict__ bias,
                          const float* __restrict__ g, const float* __restrict__ bet,
                          const float* __restrict__ mu, const float* __restrict__ var,
                          const float* __restrict__ Lprev, float* __restrict__ Lout, int total) {
    int idx = blockIdx.x * blockDim.x + threadIdx.x;
    if (idx >= total) return;
    int node = idx / 20;
    int h    = idx - node * 20;
    float pre = P[(size_t)node * 64 + h]            // p0
              + T[(size_t)node * 40 + h]            // t1 = L p1
              + 2.0f * T2[idx]                      // 2 * L(L p2)
              - P[(size_t)node * 64 + 40 + h]       // - p2
              + bias[h];
    float yv = (pre - mu[h]) * rsqrtf(var[h] + 1e-5f) * g[h] + bet[h];
    yv = fmaxf(yv, 0.0f);
    if (Lprev) yv = fmaf(0.7f, Lprev[idx], yv);
    Lout[idx] = yv;
}

// ---------------------------------------------------------------------------
// Fuse (softmax weights) + final linear (H=20 -> 2)
// ---------------------------------------------------------------------------
__global__ void fuse_out(const float* __restrict__ L0, const float* __restrict__ L1,
                         const float* __restrict__ L2, const float* __restrict__ L3,
                         const float* __restrict__ fw, const float* __restrict__ ow,
                         const float* __restrict__ ob, float* __restrict__ out, int n) {
    int node = blockIdx.x * blockDim.x + threadIdx.x;
    if (node >= n) return;
    float f0 = fw[0], f1 = fw[1], f2 = fw[2], f3 = fw[3];
    float mx = fmaxf(fmaxf(f0, f1), fmaxf(f2, f3));
    float e0 = expf(f0 - mx), e1 = expf(f1 - mx), e2 = expf(f2 - mx), e3 = expf(f3 - mx);
    float inv = 1.0f / (e0 + e1 + e2 + e3);
    float w0 = e0 * inv, w1 = e1 * inv, w2 = e2 * inv, w3 = e3 * inv;
    float a0 = ob[0], a1 = ob[1];
    size_t base = (size_t)node * 20;
    #pragma unroll
    for (int h = 0; h < 20; ++h) {
        float emb = w0 * L0[base + h] + w1 * L1[base + h] + w2 * L2[base + h] + w3 * L3[base + h];
        a0 = fmaf(emb, ow[h],      a0);
        a1 = fmaf(emb, ow[20 + h], a1);
    }
    out[(size_t)node * 2]     = a0;
    out[(size_t)node * 2 + 1] = a1;
}

// ---------------------------------------------------------------------------
// Host-side orchestration
// ---------------------------------------------------------------------------
extern "C" void kernel_launch(void* const* d_in, const int* in_sizes, int n_in,
                              void* d_out, int out_size, void* d_ws, size_t ws_size,
                              hipStream_t stream) {
    const float* features = (const float*)d_in[0];
    const int*   edges    = (const int*)d_in[1];
    const float* Wmat[4]  = {(const float*)d_in[3], (const float*)d_in[5],
                             (const float*)d_in[7], (const float*)d_in[9]};
    const float* bias[4]  = {(const float*)d_in[4], (const float*)d_in[6],
                             (const float*)d_in[8], (const float*)d_in[10]};
    const float* bn_g   = (const float*)d_in[11];
    const float* bn_b   = (const float*)d_in[12];
    const float* bn_m   = (const float*)d_in[13];
    const float* bn_v   = (const float*)d_in[14];
    const float* fuse_w = (const float*)d_in[15];
    const float* out_w  = (const float*)d_in[16];
    const float* out_b  = (const float*)d_in[17];

    const int IN_DIM = 2020, H = 20;
    const int N = in_sizes[0] / IN_DIM;
    const int E = in_sizes[1] / 2;
    const int* src = edges;
    const int* dst = edges + E;

    // ---- workspace carve (256B aligned) ----
    char* wsb = (char*)d_ws;
    size_t off = 0;
    auto carve = [&](size_t bytes) -> void* {
        void* p = (void*)(wsb + off);
        off += (bytes + 255) & ~(size_t)255;
        return p;
    };
    float* dis = (float*)carve((size_t)N * 4);
    float* nrm = (float*)carve((size_t)E * 4);
    unsigned short* WtH[4];
    unsigned short* WtL[4];
    WtH[0] = (unsigned short*)carve((size_t)64 * 2048 * 2);
    WtL[0] = (unsigned short*)carve((size_t)64 * 2048 * 2);
    for (int i = 1; i < 4; ++i) {
        WtH[i] = (unsigned short*)carve((size_t)2048 * 2);
        WtL[i] = (unsigned short*)carve((size_t)2048 * 2);
    }
    float* P  = (float*)carve((size_t)N * 64 * 4);
    float* T  = (float*)carve((size_t)N * 40 * 4);
    float* T2 = (float*)carve((size_t)N * 20 * 4);
    float* L[4];
    for (int i = 0; i < 4; ++i) L[i] = (float*)carve((size_t)N * 20 * 4);
    (void)ws_size; (void)n_in; (void)out_size;

    const int TB = 256;
    auto cdiv = [](long a, long b) { return (int)((a + b - 1) / b); };

    // ---- normalization coefficients ----
    fill_f32  <<<cdiv(N, TB), TB, 0, stream>>>(dis, 0.0f, N);
    deg_count <<<cdiv(E, TB), TB, 0, stream>>>(src, dis, E);
    deg_to_dis<<<cdiv(N, TB), TB, 0, stream>>>(dis, N);
    norm_coef <<<cdiv(E, TB), TB, 0, stream>>>(src, dst, dis, nrm, E);

    // ---- weight pre-transform (bf16 hi/lo, fragment order) ----
    wt_transform<<<cdiv(64 * 2048, TB), TB, 0, stream>>>(Wmat[0], IN_DIM, WtH[0], WtL[0], 64);
    for (int i = 1; i < 4; ++i)
        wt_transform<<<cdiv(2048, TB), TB, 0, stream>>>(Wmat[i], H, WtH[i], WtL[i], 1);

    // ---- 4 Cheb layers ----
    for (int i = 0; i < 4; ++i) {
        if (i == 0)
            wmma_gemm_bf16x3<<<cdiv(N, 128), 256, 0, stream>>>(
                features, IN_DIM, IN_DIM, 63, 64, WtH[0], WtL[0], P, N);
        else
            wmma_gemm_bf16x3<<<cdiv(N, 128), 256, 0, stream>>>(
                L[i - 1], H, H, 0, 1, WtH[i], WtL[i], P, N);

        // t1 = L p1 and u = L p2 fused into one 40-wide prop (P cols 20..59 -> T cols 0..39)
        fill_f32   <<<cdiv((long)N * 40, TB), TB, 0, stream>>>(T, 0.0f, N * 40);
        prop_scatter<<<cdiv(E, TB), TB, 0, stream>>>(P + 20, 64, T, 40, 40, src, dst, nrm, E);
        // t2 = L u   (T cols 20..39 -> T2)
        fill_f32   <<<cdiv((long)N * 20, TB), TB, 0, stream>>>(T2, 0.0f, N * 20);
        prop_scatter<<<cdiv(E, TB), TB, 0, stream>>>(T + 20, 40, T2, 20, 20, src, dst, nrm, E);

        cheb_post<<<cdiv((long)N * 20, TB), TB, 0, stream>>>(
            P, T, T2, bias[i],
            bn_g + 20 * i, bn_b + 20 * i, bn_m + 20 * i, bn_v + 20 * i,
            i ? L[i - 1] : nullptr, L[i], N * 20);
    }

    // ---- fuse + output projection ----
    fuse_out<<<cdiv(N, TB), TB, 0, stream>>>(L[0], L[1], L[2], L[3],
                                             fuse_w, out_w, out_b, (float*)d_out, N);
}
